// ConvCRF_43628277793532
// MI455X (gfx1250) — matrix operand order repaired and checked by reading, not compiled, not run on hardware
//
#include <hip/hip_runtime.h>
#include <hip/hip_bf16.h>

typedef __attribute__((ext_vector_type(16))) _Float16 v16h;
typedef __attribute__((ext_vector_type(8)))  float    v8f;

#define BM 128
#define BN 128
#define BK 32
#define LDA 40   // padded LDS row stride (halfs); 80B keeps 16B alignment of 32B runs
#define LDB 40
#define LBL 48
#define NBATCH 32

// gfx1250 async global->LDS copy path (ASYNCcnt), guarded so compile never regresses
#if defined(__HIP_DEVICE_COMPILE__) && defined(__gfx1250__)
#if __has_builtin(__builtin_amdgcn_global_load_async_to_lds_b128) && \
    __has_builtin(__builtin_amdgcn_s_wait_asynccnt)
#define ASYNC_COPY 1
#endif
#endif

#ifdef ASYNC_COPY
#define AS1 __attribute__((address_space(1)))
#define AS3 __attribute__((address_space(3)))
typedef int b128_t __attribute__((vector_size(4 * sizeof(int))));
static __device__ __forceinline__ void async_cp16(const _Float16* g, _Float16* l) {
  // moves 16 bytes (8 halfs) per lane, LDS write tracked by ASYNCcnt
  __builtin_amdgcn_global_load_async_to_lds_b128((AS1 b128_t*)g, (AS3 b128_t*)l, 0, 0);
}
static __device__ __forceinline__ void async_cp32(const _Float16* g, _Float16* l) {
  __builtin_amdgcn_global_load_async_to_lds_b128((AS1 b128_t*)g, (AS3 b128_t*)l, 0, 0);
  __builtin_amdgcn_global_load_async_to_lds_b128((AS1 b128_t*)(g + 8), (AS3 b128_t*)(l + 8), 0, 0);
}
#endif

// ---------------------------------------------------------------------------
// Transpose fp32 (rows x cols) -> f16 (cols x rows), batched over nmat slices.
// Gives BT in N x K row-major so GEMM B-fragments are contiguous LDS reads.
// ---------------------------------------------------------------------------
__global__ void transpose_f32_to_f16T(const float* __restrict__ W,
                                      _Float16* __restrict__ WT,
                                      int rows, int cols, int nmat) {
  long idx = (long)blockIdx.x * blockDim.x + threadIdx.x;
  long per = (long)rows * cols;
  long total = per * nmat;
  if (idx >= total) return;
  long m = idx / per;
  long r = idx % per;        // r = n*rows + k in destination
  long n = r / rows;
  long k = r % rows;
  WT[idx] = (_Float16)W[m * per + k * (long)cols + n];
}

// ---------------------------------------------------------------------------
// Tiled WMMA GEMM: C(f16) = act( sum_taps shift(A) x BT_tap^T + bias ) [+ resid]
//   A      : M x K row-major (fp32 or f16)
//   BT     : NTAPS x (N x K) row-major f16 (pre-transposed weights)
//   taps   : tap t reads A rows shifted by (t - NTAPS/2)*tapShift, OOB rows = 0
//            (implements SAME-padded conv1d along seq for (S,B,E) flattening)
// Block = 256 threads = 8 waves; block tile 128x128; wave tile 32x64 (2x4 WMMA)
// f16-A path: double-buffered LDS with async global->LDS copies (ASYNCcnt),
// overlapping the next chunk's fill with the current chunk's WMMAs.
// ---------------------------------------------------------------------------
template<typename TA, int NTAPS, bool RELU_RES>
__global__ __launch_bounds__(256)
void wmma_gemm_kernel(const TA* __restrict__ A,
                      const _Float16* __restrict__ BT,
                      const float* __restrict__ bias,
                      const _Float16* __restrict__ resid,
                      _Float16* __restrict__ C,
                      int M, int N, int K, int tapShift)
{
  __shared__ _Float16 As[2][BM * LDA];
  __shared__ _Float16 Bs[2][BN * LDB];
  const int tid  = threadIdx.x;
  const int lane = tid & 31;
  const int wid  = tid >> 5;
  const int wm   = wid & 3;    // 4 row-waves  -> rows wm*32 .. +32
  const int wn   = wid >> 2;   // 2 col-waves  -> cols wn*64 .. +64
  const int rowBase = blockIdx.x * BM;
  const int colBase = blockIdx.y * BN;
  const int lrow  = lane & 15;
  const int khalf = lane >> 4;

  v8f acc[2][4];
  #pragma unroll
  for (int i = 0; i < 2; ++i)
    #pragma unroll
    for (int j = 0; j < 4; ++j)
      #pragma unroll
      for (int e = 0; e < 8; ++e) acc[i][j][e] = 0.f;

#ifdef ASYNC_COPY
  if (sizeof(TA) == 2) {
    // ---------------- async double-buffered pipeline (f16 A) ----------------
    const int nkc = K / BK;
    const int nch = NTAPS * nkc;
    const int r   = tid >> 1;
    const int ko  = (tid & 1) * 16;
    const _Float16* Ah = (const _Float16*)A;

    auto stage = [&](int c, int buf) {
      int tap = c / nkc;
      int kc  = (c % nkc) * BK;
      long shift = (long)(tap - (NTAPS >> 1)) * tapShift;
      // A tile (128 x 32) with row-shift + zero padding at seq boundaries
      {
        long grow = (long)rowBase + r + shift;
        _Float16* dst = &As[buf][r * LDA + ko];
        if (grow >= 0 && grow < (long)M) {
          async_cp16(Ah + grow * (long)K + kc + ko, dst);
          async_cp16(Ah + grow * (long)K + kc + ko + 8, dst + 8);
        } else {
          #pragma unroll
          for (int e = 0; e < 16; ++e) dst[e] = (_Float16)0.f;  // ds stores, DScnt
        }
      }
      // B tile (128 cols x 32 K) from pre-transposed weights (always in range)
      async_cp32(BT + (size_t)tap * N * K + (size_t)(colBase + r) * K + kc + ko,
                 &Bs[buf][r * LDB + ko]);
    };

    stage(0, 0);
    for (int c = 0; c < nch; ++c) {
      const int buf = c & 1;
      __builtin_amdgcn_s_wait_asynccnt(0);   // my async fills for chunk c done
      __syncthreads();                       // everyone's chunk c staged; buf^1 free
      if (c + 1 < nch) stage(c + 1, buf ^ 1);

      const _Float16* Asb = As[buf];
      const _Float16* Bsb = Bs[buf];
      v16h af[2], bf[4];
      #pragma unroll
      for (int mt = 0; mt < 2; ++mt) {
        const _Float16* p = Asb + (wm * 32 + mt * 16 + lrow) * LDA;
        #pragma unroll
        for (int e = 0; e < 16; ++e) {
          // A 16x32 f16: lanes 0-15 K=0..7,16..23; lanes 16-31 K=8..15,24..31
          int kk = ((e & 8) << 1) + khalf * 8 + (e & 7);
          af[mt][e] = p[kk];
        }
      }
      #pragma unroll
      for (int nt = 0; nt < 4; ++nt) {
        // B 32x16 f16: lane = col, lanes 0-15 K=0..7,16..23; lanes 16-31 K=8..15,24..31
        const _Float16* p = Bsb + (wn * 64 + nt * 16 + lrow) * LDB;
        #pragma unroll
        for (int e = 0; e < 16; ++e) bf[nt][e] = p[((e & 8) << 1) + khalf * 8 + (e & 7)];
      }
      #pragma unroll
      for (int mt = 0; mt < 2; ++mt)
        #pragma unroll
        for (int nt = 0; nt < 4; ++nt)
          acc[mt][nt] = __builtin_amdgcn_wmma_f32_16x16x32_f16(
              false, af[mt], false, bf[nt], (short)0, acc[mt][nt], false, false);
      // no trailing barrier: next iteration's barrier separates our reads of
      // buf from the stage of chunk c+2 into buf
    }
  } else
#endif
  {
    // ---------------- synchronous single-buffer path (fp32 A or no async) ---
    for (int tap = 0; tap < NTAPS; ++tap) {
      const long shift = (long)(tap - (NTAPS >> 1)) * tapShift;
      const _Float16* BTt = BT + (size_t)tap * N * K;

      for (int kc = 0; kc < K; kc += BK) {
        {
          int r  = tid >> 1;
          int ko = (tid & 1) * 16;
          long grow = (long)rowBase + r + shift;
          _Float16* dst = &As[0][r * LDA + ko];
          if (grow >= 0 && grow < (long)M) {
            const TA* src = A + grow * (long)K + kc + ko;
            #pragma unroll
            for (int e = 0; e < 16; ++e) dst[e] = (_Float16)src[e];
          } else {
            #pragma unroll
            for (int e = 0; e < 16; ++e) dst[e] = (_Float16)0.f;
          }
        }
        {
          int r  = tid >> 1;
          int ko = (tid & 1) * 16;
          const _Float16* src = BTt + (size_t)(colBase + r) * K + kc + ko;
          _Float16* dst = &Bs[0][r * LDB + ko];
          #pragma unroll
          for (int e = 0; e < 16; ++e) dst[e] = src[e];
        }
        __syncthreads();

        v16h af[2], bf[4];
        #pragma unroll
        for (int mt = 0; mt < 2; ++mt) {
          const _Float16* p = &As[0][(wm * 32 + mt * 16 + lrow) * LDA];
          #pragma unroll
          for (int e = 0; e < 16; ++e) {
            int kk = ((e & 8) << 1) + khalf * 8 + (e & 7);
            af[mt][e] = p[kk];
          }
        }
        #pragma unroll
        for (int nt = 0; nt < 4; ++nt) {
          const _Float16* p = &Bs[0][(wn * 64 + nt * 16 + lrow) * LDB];
          #pragma unroll
          for (int e = 0; e < 16; ++e) bf[nt][e] = p[((e & 8) << 1) + khalf * 8 + (e & 7)];
        }
        #pragma unroll
        for (int mt = 0; mt < 2; ++mt)
          #pragma unroll
          for (int nt = 0; nt < 4; ++nt)
            acc[mt][nt] = __builtin_amdgcn_wmma_f32_16x16x32_f16(
                false, af[mt], false, bf[nt], (short)0, acc[mt][nt], false, false);
        __syncthreads();
      }
    }
  }

  // ---- epilogue: C/D layout VGPR i -> M = i + 8*lane[4], N = lane[3:0] ----
  const int erow = lane >> 4;
  #pragma unroll
  for (int mt = 0; mt < 2; ++mt) {
    #pragma unroll
    for (int nt = 0; nt < 4; ++nt) {
      int gcol = colBase + wn * 64 + nt * 16 + lrow;
      float bv = bias ? bias[gcol] : 0.f;
      #pragma unroll
      for (int i = 0; i < 8; ++i) {
        long grow = (long)rowBase + wm * 32 + mt * 16 + erow * 8 + i;
        float v = acc[mt][nt][i] + bv;
        if (RELU_RES) v = fmaxf(v, 0.f) + (float)resid[grow * (long)N + gcol];
        C[grow * (long)N + gcol] = (_Float16)v;
      }
    }
  }
}

// ---------------------------------------------------------------------------
// Narrow-N out-projection: feats(f32, M x 48) = A(f16, M x K) x Wout^T + b
// 8 waves, each owns a 16-row strip of the 128-row tile, 3 WMMA N-tiles.
// ---------------------------------------------------------------------------
__global__ __launch_bounds__(256)
void wmma_outproj_kernel(const _Float16* __restrict__ A,
                         const _Float16* __restrict__ BT,  // 48 x K
                         const float* __restrict__ bias,   // 48
                         float* __restrict__ C,            // M x 48
                         int M, int K)
{
  __shared__ _Float16 As[BM * LDA];
  __shared__ _Float16 Bs[LBL * LDB];
  const int tid   = threadIdx.x;
  const int lane  = tid & 31;
  const int wid   = tid >> 5;
  const int rowBase = blockIdx.x * BM;
  const int lrow  = lane & 15;
  const int khalf = lane >> 4;

  v8f acc[3];
  #pragma unroll
  for (int j = 0; j < 3; ++j)
    #pragma unroll
    for (int e = 0; e < 8; ++e) acc[j][e] = 0.f;

  for (int kc = 0; kc < K; kc += BK) {
    {
      int r = tid >> 1; int ko = (tid & 1) * 16;
      const _Float16* src = A + ((long)rowBase + r) * K + kc + ko;
      _Float16* dst = As + r * LDA + ko;
      #pragma unroll
      for (int e = 0; e < 16; ++e) dst[e] = src[e];
    }
    if (tid < 96) {  // 48 rows x 32 halfs
      int r = tid >> 1; int ko = (tid & 1) * 16;
      const _Float16* src = BT + (long)r * K + kc + ko;
      _Float16* dst = Bs + r * LDB + ko;
      #pragma unroll
      for (int e = 0; e < 16; ++e) dst[e] = src[e];
    }
    __syncthreads();

    v16h af, bf[3];
    {
      const _Float16* p = As + (wid * 16 + lrow) * LDA;
      #pragma unroll
      for (int e = 0; e < 16; ++e) af[e] = p[((e & 8) << 1) + khalf * 8 + (e & 7)];
    }
    #pragma unroll
    for (int nt = 0; nt < 3; ++nt) {
      const _Float16* p = Bs + (nt * 16 + lrow) * LDB;
      #pragma unroll
      for (int e = 0; e < 16; ++e) bf[nt][e] = p[((e & 8) << 1) + khalf * 8 + (e & 7)];
    }
    #pragma unroll
    for (int nt = 0; nt < 3; ++nt)
      acc[nt] = __builtin_amdgcn_wmma_f32_16x16x32_f16(
          false, af, false, bf[nt], (short)0, acc[nt], false, false);
    __syncthreads();
  }

  const int erow = lane >> 4;
  #pragma unroll
  for (int nt = 0; nt < 3; ++nt) {
    int gcol = nt * 16 + lrow;
    float bv = bias[gcol];
    #pragma unroll
    for (int i = 0; i < 8; ++i) {
      long grow = (long)rowBase + wid * 16 + erow * 8 + i;
      C[grow * LBL + gcol] = acc[nt][i] + bv;
    }
  }
}

// ---------------------------------------------------------------------------
// Viterbi: single block, LDS-resident state, sequential over S with barriers,
// then 32-lane backtrace. out[0:32]=score, out[32: ]=labels (float32).
// First-max tie-break (strict >) matches jnp.argmax.
// ---------------------------------------------------------------------------
__global__ __launch_bounds__(768)
void viterbi_kernel(const float* __restrict__ feats,   // (S, 32, 48)
                    const float* __restrict__ trans,   // (48, 48) [next][prev]
                    unsigned char* __restrict__ bp,    // (S, 32, 48)
                    float* __restrict__ out,
                    int S)
{
  __shared__ float T[LBL * LBL];
  __shared__ float vbuf[2][NBATCH * LBL];
  const int tid = threadIdx.x;
  for (int i = tid; i < LBL * LBL; i += blockDim.x) T[i] = trans[i];
  for (int i = tid; i < NBATCH * LBL; i += blockDim.x)
    vbuf[0][i] = ((i % LBL) == 0) ? 0.f : -1e4f;   // START=0
  __syncthreads();

  int cur = 0;
  for (int t = 0; t < S; ++t) {
    const float* ft = feats + (long)t * NBATCH * LBL;
    unsigned char* bpt = bp + (long)t * NBATCH * LBL;
    const float* vc = vbuf[cur];
    float* vn = vbuf[cur ^ 1];
    for (int i = tid; i < NBATCH * LBL; i += blockDim.x) {
      int b = i / LBL, nxt = i % LBL;
      const float* vb = vc + b * LBL;
      const float* Tr = T + nxt * LBL;
      float best = vb[0] + Tr[0];
      int bi = 0;
      #pragma unroll
      for (int p = 1; p < LBL; ++p) {
        float sc = vb[p] + Tr[p];
        if (sc > best) { best = sc; bi = p; }
      }
      vn[i] = best + ft[i];
      bpt[i] = (unsigned char)bi;
    }
    cur ^= 1;
    __syncthreads();
  }

  if (tid < NBATCH) {
    const float* vf = vbuf[cur];
    int b = tid;
    float best = vf[b * LBL] + T[1 * LBL];        // STOP=1
    int bi = 0;
    for (int l = 1; l < LBL; ++l) {
      float sc = vf[b * LBL + l] + T[1 * LBL + l];
      if (sc > best) { best = sc; bi = l; }
    }
    out[b] = best;
    int tag = bi;
    for (int t = S - 1; t >= 0; --t) {
      out[NBATCH + (long)t * NBATCH + b] = (float)tag;
      tag = bp[(long)t * NBATCH * LBL + (long)b * LBL + tag];
    }
  }
}

// ---------------------------------------------------------------------------
extern "C" void kernel_launch(void* const* d_in, const int* in_sizes, int n_in,
                              void* d_out, int out_size, void* d_ws, size_t ws_size,
                              hipStream_t stream)
{
  const float* source = (const float*)d_in[0];  // (2048, 32, 512)
  const float* W_in   = (const float*)d_in[1];  // (512, 512)
  const float* b_in   = (const float*)d_in[2];  // (512,)
  const float* conv_w = (const float*)d_in[3];  // (3, 3, 512, 512)
  const float* conv_b = (const float*)d_in[4];  // (3, 512)
  const float* W_out  = (const float*)d_in[5];  // (512, 48)
  const float* b_out  = (const float*)d_in[6];  // (48,)
  const float* trans  = (const float*)d_in[7];  // (48, 48)

  const int S = 2048, B = 32, D = 512, E = 512, L = LBL;
  const long M = (long)S * B;  // 65536

  char* ws = (char*)d_ws;
  size_t off = 0;
  auto carve = [&](size_t bytes) -> void* {
    void* p = ws + off;
    off += (bytes + 255) & ~(size_t)255;
    return p;
  };

  _Float16* hWin  = (_Float16*)carve((size_t)E * D * 2);        // 512 KB
  _Float16* hConv = (_Float16*)carve((size_t)9 * E * E * 2);    // 4.5 MB
  _Float16* hWout = (_Float16*)carve((size_t)L * E * 2);        // 48 KB
  _Float16* x0    = (_Float16*)carve((size_t)M * E * 2);        // 64 MB
  _Float16* x1    = (_Float16*)carve((size_t)M * E * 2);        // 64 MB
  float*    feats = (float*)   carve((size_t)M * L * 4);        // 12 MB
  unsigned char* bp = (unsigned char*)carve((size_t)M * L);     // 3 MB

  // Pre-transpose + f16-cast all weight matrices (B operands become N x K).
  { long tot = (long)D * E;
    transpose_f32_to_f16T<<<(tot + 255) / 256, 256, 0, stream>>>(W_in, hWin, D, E, 1); }
  { long tot = (long)9 * E * E;
    transpose_f32_to_f16T<<<(tot + 255) / 256, 256, 0, stream>>>(conv_w, hConv, E, E, 9); }
  { long tot = (long)E * L;
    transpose_f32_to_f16T<<<(tot + 255) / 256, 256, 0, stream>>>(W_out, hWout, E, L, 1); }

  dim3 g((unsigned)(M / BM), (unsigned)(E / BN));  // (512, 4)

  // in_proj: x0 = source @ W_in + b_in  (fp32 A, f16 out)
  wmma_gemm_kernel<float, 1, false><<<g, 256, 0, stream>>>(
      source, hWin, b_in, nullptr, x0, (int)M, E, D, 0);

  // residual conv blocks: x = relu(sum_k shift(x) @ W_k + b) + x
  _Float16* xin = x0; _Float16* xout = x1;
  for (int blk = 0; blk < 3; ++blk) {
    wmma_gemm_kernel<_Float16, 3, true><<<g, 256, 0, stream>>>(
        xin, hConv + (size_t)blk * 3 * E * E, conv_b + (size_t)blk * E,
        xin, xout, (int)M, E, E, B);
    _Float16* t = xin; xin = xout; xout = t;
  }

  // out_proj -> fp32 emissions
  wmma_outproj_kernel<<<dim3((unsigned)(M / BM)), 256, 0, stream>>>(
      xin, hWout, b_out, feats, (int)M, E);

  // Viterbi forward + backtrace
  viterbi_kernel<<<1, 768, 0, stream>>>(feats, trans, bp, (float*)d_out, S);
}
